// FiLMCondTwoLayerPreActivationResNetLinear_76768245448977
// MI455X (gfx1250) — matrix operand order repaired and checked by reading, not compiled, not run on hardware
//
#include <hip/hip_runtime.h>

// ---------------------------------------------------------------------------
// FiLM-conditioned two-layer pre-activation ResNet block, fused for gfx1250.
//   gb    = cond @ W_film + b_film ; gamma,beta = split(gb)
//   h0    = relu(gamma*x + beta)              (kept in LDS, bf16)
//   h1    = relu(h0 @ W1 + b1)                (kept in LDS, bf16)
//   out   = h1 @ W2 + b2 + x                  (fp32 to HBM)
// All GEMMs: V_WMMA_F32_16X16X32_BF16, fp32 accumulate.
// ---------------------------------------------------------------------------

typedef __attribute__((ext_vector_type(16))) __bf16 v16bf;
typedef __attribute__((ext_vector_type(8)))  __bf16 v8bf;
typedef __attribute__((ext_vector_type(4)))  __bf16 v4bf;
typedef __attribute__((ext_vector_type(8)))  float  v8f;

#define NROWS 65536
#define HD    256      // H
#define CD    512      // C
#define MBLK  64       // rows per workgroup

// LDS row pitches (elements), padded so 16B reads from consecutive rows hit
// different bank groups (pitch*2B is 16B-aligned, not a multiple of 256B).
#define COND_LD 520    // 512 + 8
#define H_LD    264    // 256 + 8

#define SMEM_BYTES ((MBLK * COND_LD + 2 * MBLK * H_LD) * 2)   // 134144 B

__device__ __forceinline__ v8f wmma_bf16(v16bf a, v16bf b, v8f c) {
  // (neg_a, A, neg_b, B, c_mod, C, reuse_a, reuse_b)
  return __builtin_amdgcn_wmma_f32_16x16x32_bf16(false, a, false, b,
                                                 (short)0, c, false, false);
}

// A fragment (16x32 bf16, M x K) from LDS.  ISA 7.12.2 layout:
// lane<16: row=lane,   K = {kb..kb+7, 16+kb..16+kb+7} with kb=0
// lane>=16: row=lane-16, same with kb=8
__device__ __forceinline__ v16bf load_a_frag(const __bf16* base, int pitch,
                                             int lane) {
  const int row = lane & 15;
  const int kb  = (lane >> 4) << 3;                 // 0 or 8
  union { v16bf v; v8bf h[2]; } f;
  f.h[0] = *(const v8bf*)(base + row * pitch + kb);
  f.h[1] = *(const v8bf*)(base + row * pitch + kb + 16);
  return f.v;
}

// B fragment (32x16 bf16, K x N) from global: lane holds column N=lane&15,
// K = {kb..kb+7, 16+kb..16+kb+7} with kb = 8*(lane>>4) (matches A).
__device__ __forceinline__ v16bf load_b_frag(const __bf16* base, int ld,
                                             int lane) {
  const int n  = lane & 15;
  const int kb = (lane >> 4) << 3;
  v16bf f;
#pragma unroll
  for (int s = 0; s < 8; ++s) {
    f[s]     = base[(size_t)(kb + s) * ld + n];
    f[s + 8] = base[(size_t)(kb + 16 + s) * ld + n];
  }
  return f;
}

__global__ void __launch_bounds__(256)
film_resnet_fused(const float* __restrict__ x, const float* __restrict__ cond,
                  const __bf16* __restrict__ Wf, const float* __restrict__ bfilm,
                  const __bf16* __restrict__ W1, const float* __restrict__ b1,
                  const __bf16* __restrict__ W2, const float* __restrict__ b2,
                  float* __restrict__ out) {
  extern __shared__ __bf16 smem[];
  __bf16* s_cond = smem;                        // [MBLK][COND_LD]
  __bf16* s_h0   = smem + MBLK * COND_LD;       // [MBLK][H_LD]
  __bf16* s_h1   = s_h0 + MBLK * H_LD;          // [MBLK][H_LD]

  const int tid    = threadIdx.x;
  const int lane   = tid & 31;
  const int wave   = tid >> 5;
  const int mg     = wave & 1;    // m-group: m-tiles {2mg, 2mg+1}
  const int ng     = wave >> 1;   // n-group: 0..3 (64 columns each)
  const int nlo    = lane & 15;   // column within 16-wide tile (C/D layout)
  const int rhi    = (lane >> 4) << 3;   // row offset 0 / 8 (C/D layout)
  const int base_m = blockIdx.x * MBLK;

  // ---- stage cond tile: fp32 global -> bf16 LDS (coalesced float4) --------
  for (int i = tid; i < MBLK * (CD / 4); i += 256) {
    const int row = i >> 7;           // CD/4 = 128 float4 per row
    const int c4  = i & 127;
    const float4 v =
        *(const float4*)(cond + (size_t)(base_m + row) * CD + c4 * 4);
    v4bf o;
    o[0] = (__bf16)v.x; o[1] = (__bf16)v.y;
    o[2] = (__bf16)v.z; o[3] = (__bf16)v.w;
    *(v4bf*)(s_cond + row * COND_LD + c4 * 4) = o;
  }
  __syncthreads();

  // ==== GEMM A: gb = cond @ W_film ; FiLM epilogue -> s_h0 =================
  {
    v8f accG[2][4] = {};   // gamma columns ng*64 + j*16
    v8f accB[2][4] = {};   // beta  columns 256 + ng*64 + j*16
    for (int kt = 0; kt < CD / 32; ++kt) {
      const __bf16* a_base = s_cond + kt * 32;
      const v16bf a0 = load_a_frag(a_base + (mg * 32) * COND_LD, COND_LD, lane);
      const v16bf a1 = load_a_frag(a_base + (mg * 32 + 16) * COND_LD, COND_LD, lane);
      const __bf16* wrow = Wf + (size_t)(kt * 32) * (2 * HD);
      __builtin_prefetch(wrow + (size_t)32 * (2 * HD) + (size_t)lane * (2 * HD)
                              + ng * 64, 0, 0);
#pragma unroll
      for (int j = 0; j < 4; ++j) {
        const v16bf bG = load_b_frag(wrow + ng * 64 + j * 16, 2 * HD, lane);
        accG[0][j] = wmma_bf16(a0, bG, accG[0][j]);
        accG[1][j] = wmma_bf16(a1, bG, accG[1][j]);
        const v16bf bB = load_b_frag(wrow + HD + ng * 64 + j * 16, 2 * HD, lane);
        accB[0][j] = wmma_bf16(a0, bB, accB[0][j]);
        accB[1][j] = wmma_bf16(a1, bB, accB[1][j]);
      }
    }
#pragma unroll
    for (int mi = 0; mi < 2; ++mi) {
#pragma unroll
      for (int j = 0; j < 4; ++j) {
        const int col = ng * 64 + j * 16 + nlo;
        const float gbias = bfilm[col];
        const float bbias = bfilm[HD + col];
#pragma unroll
        for (int r = 0; r < 8; ++r) {
          const int rowb = (mg * 2 + mi) * 16 + rhi + r;
          const float xv = x[(size_t)(base_m + rowb) * HD + col];
          float hv = (accG[mi][j][r] + gbias) * xv + (accB[mi][j][r] + bbias);
          hv = fmaxf(hv, 0.0f);
          s_h0[rowb * H_LD + col] = (__bf16)hv;
        }
      }
    }
  }
  __syncthreads();

  // ==== GEMM B: h1 = relu(h0 @ W1 + b1) -> s_h1 ============================
  {
    v8f acc[2][4] = {};
    for (int kt = 0; kt < HD / 32; ++kt) {
      const __bf16* a_base = s_h0 + kt * 32;
      const v16bf a0 = load_a_frag(a_base + (mg * 32) * H_LD, H_LD, lane);
      const v16bf a1 = load_a_frag(a_base + (mg * 32 + 16) * H_LD, H_LD, lane);
      const __bf16* wrow = W1 + (size_t)(kt * 32) * HD;
      __builtin_prefetch(wrow + (size_t)32 * HD + (size_t)lane * HD + ng * 64,
                         0, 0);
#pragma unroll
      for (int j = 0; j < 4; ++j) {
        const v16bf b = load_b_frag(wrow + ng * 64 + j * 16, HD, lane);
        acc[0][j] = wmma_bf16(a0, b, acc[0][j]);
        acc[1][j] = wmma_bf16(a1, b, acc[1][j]);
      }
    }
#pragma unroll
    for (int mi = 0; mi < 2; ++mi) {
#pragma unroll
      for (int j = 0; j < 4; ++j) {
        const int col  = ng * 64 + j * 16 + nlo;
        const float bb = b1[col];
#pragma unroll
        for (int r = 0; r < 8; ++r) {
          const int rowb = (mg * 2 + mi) * 16 + rhi + r;
          const float hv = fmaxf(acc[mi][j][r] + bb, 0.0f);
          s_h1[rowb * H_LD + col] = (__bf16)hv;
        }
      }
    }
  }
  __syncthreads();

  // ==== GEMM C: out = h1 @ W2 + b2 + x =====================================
  {
    v8f acc[2][4] = {};
    for (int kt = 0; kt < HD / 32; ++kt) {
      const __bf16* a_base = s_h1 + kt * 32;
      const v16bf a0 = load_a_frag(a_base + (mg * 32) * H_LD, H_LD, lane);
      const v16bf a1 = load_a_frag(a_base + (mg * 32 + 16) * H_LD, H_LD, lane);
      const __bf16* wrow = W2 + (size_t)(kt * 32) * HD;
#pragma unroll
      for (int j = 0; j < 4; ++j) {
        const v16bf b = load_b_frag(wrow + ng * 64 + j * 16, HD, lane);
        acc[0][j] = wmma_bf16(a0, b, acc[0][j]);
        acc[1][j] = wmma_bf16(a1, b, acc[1][j]);
      }
    }
#pragma unroll
    for (int mi = 0; mi < 2; ++mi) {
#pragma unroll
      for (int j = 0; j < 4; ++j) {
        const int col  = ng * 64 + j * 16 + nlo;
        const float bb = b2[col];
#pragma unroll
        for (int r = 0; r < 8; ++r) {
          const int rowb = (mg * 2 + mi) * 16 + rhi + r;
          const size_t gidx = (size_t)(base_m + rowb) * HD + col;
          out[gidx] = acc[mi][j][r] + bb + x[gidx];
        }
      }
    }
  }
}

// fp32 -> bf16 weight conversion (weights are tiny; runs once per launch)
__global__ void __launch_bounds__(256)
cvt_f32_bf16(const float* __restrict__ src, __bf16* __restrict__ dst, int n4) {
  const int i = blockIdx.x * blockDim.x + threadIdx.x;
  if (i < n4) {
    const float4 v = ((const float4*)src)[i];
    v4bf o;
    o[0] = (__bf16)v.x; o[1] = (__bf16)v.y;
    o[2] = (__bf16)v.z; o[3] = (__bf16)v.w;
    ((v4bf*)dst)[i] = o;
  }
}

extern "C" void kernel_launch(void* const* d_in, const int* in_sizes, int n_in,
                              void* d_out, int out_size, void* d_ws,
                              size_t ws_size, hipStream_t stream) {
  (void)in_sizes; (void)n_in; (void)out_size; (void)ws_size;
  const float* x     = (const float*)d_in[0];
  const float* cond  = (const float*)d_in[1];
  const float* Wf    = (const float*)d_in[2];
  const float* bfilm = (const float*)d_in[3];
  const float* W1    = (const float*)d_in[4];
  const float* b1    = (const float*)d_in[5];
  const float* W2    = (const float*)d_in[6];
  const float* b2    = (const float*)d_in[7];
  float* out = (float*)d_out;

  __bf16* Wf_b = (__bf16*)d_ws;                 // 512*512 bf16
  __bf16* W1_b = Wf_b + CD * (2 * HD);          // 256*256 bf16
  __bf16* W2_b = W1_b + HD * HD;

  cvt_f32_bf16<<<(CD * 2 * HD / 4 + 255) / 256, 256, 0, stream>>>(Wf, Wf_b,
                                                                  CD * 2 * HD / 4);
  cvt_f32_bf16<<<(HD * HD / 4 + 255) / 256, 256, 0, stream>>>(W1, W1_b,
                                                              HD * HD / 4);
  cvt_f32_bf16<<<(HD * HD / 4 + 255) / 256, 256, 0, stream>>>(W2, W2_b,
                                                              HD * HD / 4);

  hipFuncSetAttribute((const void*)film_resnet_fused,
                      hipFuncAttributeMaxDynamicSharedMemorySize, SMEM_BYTES);

  film_resnet_fused<<<NROWS / MBLK, 256, SMEM_BYTES, stream>>>(
      x, cond, Wf_b, bfilm, W1_b, b1, W2_b, b2, out);
}
